// Encoder_40218073760118
// MI455X (gfx1250) — matrix operand order repaired and checked by reading, not compiled, not run on hardware
//
#include <hip/hip_runtime.h>
#include <hip/hip_bf16.h>

// ---- problem dimensions (from reference) ----
#define NB   2048
#define N_   64
#define F_   32
#define NE   4        // E-1 used edge types
#define U1_  128
#define U2_  64
#define UA_  128
#define AN1  (U1_ + F_)   // 160
#define AN2  (U2_ + F_)   // 96

typedef __attribute__((ext_vector_type(16))) __bf16         v16bf;
typedef __attribute__((ext_vector_type(16))) unsigned short v16u;
typedef __attribute__((ext_vector_type(8)))  unsigned short v8u;
typedef __attribute__((ext_vector_type(8)))  float          v8f;

// ---- packed-weight workspace layout (ushort offsets), all [N][K] transposed bf16 ----
#define WS_W1T  0                          // [4][128][32]
#define WS_V1T  (WS_W1T + NE*U1_*F_)       // [128][32]
#define WS_W2T  (WS_V1T + U1_*F_)          // [4][64][160]
#define WS_V2T  (WS_W2T + NE*U2_*AN1)      // [64][32]
#define WS_WIT  (WS_V2T + U2_*F_)          // [128][96]
#define WS_WJT  (WS_WIT + UA_*AN2)         // [128][96]
#define WS_END  (WS_WJT + UA_*AN2)         // 88064 ushorts = 176128 bytes

static __device__ __forceinline__ unsigned short f2bfu(float f) {
    unsigned u = __builtin_bit_cast(unsigned, f);
    return (unsigned short)(u >> 16);     // truncate to bf16
}

static __device__ __forceinline__ v16bf frag_cat(v8u lo, v8u hi) {
    v16u t = __builtin_shufflevector(lo, hi, 0, 1, 2, 3, 4, 5, 6, 7,
                                             8, 9, 10, 11, 12, 13, 14, 15);
    return __builtin_bit_cast(v16bf, t);
}

// A-fragment (16x32 bf16) from row-major [M][K] source, stride ld (elements).
// Lane L: M = L%16, half = L/16. Values: K = half*8 + {0..7} and {16..23} -> two b128 loads.
static __device__ __forceinline__ v16bf a_frag(const unsigned short* base,
                                               int row0, int col0, int ld, int lane) {
    const unsigned short* r = base + (row0 + (lane & 15)) * ld + col0 + ((lane >> 4) << 3);
    v8u lo = *(const v8u*)r;
    v8u hi = *(const v8u*)(r + 16);
    return frag_cat(lo, hi);
}

// B-fragment (32x16 bf16) from TRANSPOSED [N][K] source, stride ldK (elements).
// Lane L: N = L%16, kh = L/16. Values: K = kh*8 + {0..7} and {16..23} -> two b128 loads.
static __device__ __forceinline__ v16bf b_fragT(const unsigned short* baseT,
                                                int k0, int n0, int ldK, int lane) {
    const unsigned short* r = baseT + (n0 + (lane & 15)) * ldK + k0 + ((lane >> 4) << 3);
    v8u lo = *(const v8u*)r;
    v8u hi = *(const v8u*)(r + 16);
    return frag_cat(lo, hi);
}

static __device__ __forceinline__ v8f wmma_bf16(v16bf a, v16bf b, v8f c) {
    return __builtin_amdgcn_wmma_f32_16x16x32_bf16(false, a, false, b, (short)0, c,
                                                   false, false);
}

// Store a 16x16 f32 D-tile (+bias) transposed into LDS [U][M] as bf16: one b128 per lane.
// Lane L: u = u0 + L%16, m = m0 + (L/16)*8 + {0..7} (contiguous).
static __device__ __forceinline__ void store_tileT(unsigned short* baseT, int m0, int u0,
                                                   int ldM, v8f d, float bias, int lane) {
    v8u st;
#pragma unroll
    for (int v = 0; v < 8; ++v) st[v] = f2bfu(d[v] + bias);
    *(v8u*)(baseT + (u0 + (lane & 15)) * ldM + m0 + ((lane >> 4) << 3)) = st;
}

// Fast activations on TRANS ops (v_exp_f32 + v_rcp_f32); saturate correctly at +/-inf.
static __device__ __forceinline__ float fast_tanh(float x) {
    // tanh(x) = 1 - 2 / (1 + exp(2x))
    return 1.0f - 2.0f * __builtin_amdgcn_rcpf(1.0f + __expf(2.0f * x));
}
static __device__ __forceinline__ float fast_sigmoid(float x) {
    return __builtin_amdgcn_rcpf(1.0f + __expf(-x));
}

// ------------------------------------------------------------------
// One-shot weight pack: f32 [K][N] -> bf16 [N][K] in workspace.
// ------------------------------------------------------------------
__global__ __launch_bounds__(256)
void pack_weights_kernel(const float* __restrict__ W1, const float* __restrict__ V1,
                         const float* __restrict__ W2, const float* __restrict__ V2,
                         const float* __restrict__ Wi, const float* __restrict__ Wj,
                         unsigned short* __restrict__ ws) {
    const int idx = blockIdx.x * 256 + threadIdx.x;
    if (idx >= WS_END) return;
    float v;
    if (idx < WS_V1T) {                      // W1t[e][u][k] <- W1[e][k][u]
        const int e = idx / (U1_ * F_), r = idx % (U1_ * F_), u = r / F_, k = r % F_;
        v = W1[(e * F_ + k) * U1_ + u];
    } else if (idx < WS_W2T) {               // V1t[u][k] <- V1[k][u]
        const int i = idx - WS_V1T, u = i / F_, k = i % F_;
        v = V1[k * U1_ + u];
    } else if (idx < WS_V2T) {               // W2t[e][u][k] <- W2[e][k][u]
        const int i = idx - WS_W2T;
        const int e = i / (U2_ * AN1), r = i % (U2_ * AN1), u = r / AN1, k = r % AN1;
        v = W2[(e * AN1 + k) * U2_ + u];
    } else if (idx < WS_WIT) {               // V2t[u][k] <- V2[k][u]
        const int i = idx - WS_V2T, u = i / F_, k = i % F_;
        v = V2[k * U2_ + u];
    } else if (idx < WS_WJT) {               // Wit[u][k] <- Wi[k][u]
        const int i = idx - WS_WIT, u = i / AN2, k = i % AN2;
        v = Wi[k * UA_ + u];
    } else {                                 // Wjt[u][k] <- Wj[k][u]
        const int i = idx - WS_WJT, u = i / AN2, k = i % AN2;
        v = Wj[k * UA_ + u];
    }
    ws[idx] = f2bfu(v);
}

// ------------------------------------------------------------------
// Main encoder kernel: one 256-thread block (8 wave32) per batch element.
// ------------------------------------------------------------------
__global__ __launch_bounds__(256)
void encoder_ggnn_kernel(const float* __restrict__ adjacency,        // [B,N,N,5]
                         const float* __restrict__ node,             // [B,N,F]
                         const unsigned short* __restrict__ wp,      // packed bf16 weights
                         const float* __restrict__ b1,               // [4,U1]
                         const float* __restrict__ c1,               // [U1]
                         const float* __restrict__ b2,               // [4,U2]
                         const float* __restrict__ c2,               // [U2]
                         const float* __restrict__ bi,               // [UA]
                         const float* __restrict__ bj,               // [UA]
                         float* __restrict__ out)                    // [B,UA]
{
    extern __shared__ unsigned char smem_raw[];
    unsigned short* s_adj   = (unsigned short*)smem_raw;         // [4][64][64] row-major (A)
    unsigned short* s_node  = s_adj  + NE * N_ * N_;             // [64][32]    row-major (A)
    unsigned short* s_ann1  = s_node + N_ * F_;                  // [64][160] = [h1|node] (A)
    unsigned short* s_ann2  = s_ann1 + N_ * AN1;                 // [64][96]  = [h2|node] (A)
    unsigned short* s_ptmpT = s_ann2 + N_ * AN2;                 // [128][64] TRANSPOSED (B)
    float*          s_red   = (float*)(s_ptmpT + U1_ * N_);      // [128]

    const int b    = blockIdx.x;
    const int tid  = threadIdx.x;
    const int lane = tid & 31;
    const int wave = tid >> 5;
    const int n16  = lane & 15;

    const float* adjB  = adjacency + (size_t)b * N_ * N_ * 5;
    const float* nodeB = node + (size_t)b * N_ * F_;

    // Warm the adjacency slab (global_prefetch_b8)
    __builtin_prefetch(adjB + (size_t)tid * 80, 0, 1);

    // ---------------- Phase 0: stage node + adjacency (bf16) into LDS ----------------
    for (int i = tid; i < N_ * F_; i += 256) {
        const unsigned short h = f2bfu(nodeB[i]);
        const int m = i >> 5, f = i & 31;
        s_node[i]                 = h;
        s_ann1[m * AN1 + U1_ + f] = h;
        s_ann2[m * AN2 + U2_ + f] = h;
    }
    // adj[b,m,n,1:5] is 4 contiguous floats; s_adj[e][m][n] = adj_t[b,e,m,n]
    for (int i = tid; i < N_ * N_; i += 256) {
        const float* src = adjB + (size_t)i * 5 + 1;
#pragma unroll
        for (int e = 0; e < NE; ++e) s_adj[e * (N_ * N_) + i] = f2bfu(src[e]);
    }
    if (tid < UA_) s_red[tid] = 0.0f;
    __syncthreads();

    // ---------------- Phase 1: h1 = tanh(sum_e adj_e @ (node@W1_e + b1_e) + node@V1 + c1)
    // 32 output tiles (4 mtiles x 8 ntiles), 4 per wave
    v8f acc1[4] = {};
#pragma unroll
    for (int e = 0; e < NE; ++e) {
        const unsigned short* W1t = wp + WS_W1T + e * U1_ * F_;   // [128][32]
        const float* b1e = b1 + e * U1_;
#pragma unroll
        for (int i = 0; i < 4; ++i) {
            const int t = wave * 4 + i, mt = t >> 3, nt = t & 7;
            v16bf a  = a_frag(s_node, mt * 16, 0, F_, lane);
            v16bf bb = b_fragT(W1t, 0, nt * 16, F_, lane);
            v8f p = {};
            p = wmma_bf16(a, bb, p);
            store_tileT(s_ptmpT, mt * 16, nt * 16, N_, p, b1e[nt * 16 + n16], lane);
        }
        __syncthreads();
        const unsigned short* adjE = s_adj + e * N_ * N_;
#pragma unroll
        for (int i = 0; i < 4; ++i) {
            const int t = wave * 4 + i, mt = t >> 3, nt = t & 7;
#pragma unroll
            for (int ks = 0; ks < 2; ++ks) {
                v16bf a  = a_frag(adjE, mt * 16, ks * 32, N_, lane);
                v16bf bb = b_fragT(s_ptmpT, ks * 32, nt * 16, N_, lane);
                acc1[i] = wmma_bf16(a, bb, acc1[i]);
            }
        }
        __syncthreads();
    }
#pragma unroll
    for (int i = 0; i < 4; ++i) {
        const int t = wave * 4 + i, mt = t >> 3, nt = t & 7;
        v16bf a  = a_frag(s_node, mt * 16, 0, F_, lane);
        v16bf bb = b_fragT(wp + WS_V1T, 0, nt * 16, F_, lane);
        acc1[i] = wmma_bf16(a, bb, acc1[i]);
        const int   mh8  = (lane >> 4) << 3;
        const float bias = c1[nt * 16 + n16];
        unsigned short* dst = s_ann1 + (mt * 16 + mh8) * AN1 + nt * 16 + n16;
#pragma unroll
        for (int v = 0; v < 8; ++v) dst[v * AN1] = f2bfu(fast_tanh(acc1[i][v] + bias));
    }
    __syncthreads();

    // ---------------- Phase 2: h2 = tanh(sum_e adj_e @ (ann1@W2_e + b2_e) + node@V2 + c2)
    // 16 output tiles (4 x 4), 2 per wave
    v8f acc2[2] = {};
#pragma unroll
    for (int e = 0; e < NE; ++e) {
        const unsigned short* W2t = wp + WS_W2T + e * U2_ * AN1;  // [64][160]
        const float* b2e = b2 + e * U2_;
#pragma unroll
        for (int i = 0; i < 2; ++i) {
            const int t = wave * 2 + i, mt = t >> 2, nt = t & 3;
            v8f p = {};
#pragma unroll
            for (int ks = 0; ks < 5; ++ks) {   // K = 160
                v16bf a  = a_frag(s_ann1, mt * 16, ks * 32, AN1, lane);
                v16bf bb = b_fragT(W2t, ks * 32, nt * 16, AN1, lane);
                p = wmma_bf16(a, bb, p);
            }
            store_tileT(s_ptmpT, mt * 16, nt * 16, N_, p, b2e[nt * 16 + n16], lane);
        }
        __syncthreads();
        const unsigned short* adjE = s_adj + e * N_ * N_;
#pragma unroll
        for (int i = 0; i < 2; ++i) {
            const int t = wave * 2 + i, mt = t >> 2, nt = t & 3;
#pragma unroll
            for (int ks = 0; ks < 2; ++ks) {
                v16bf a  = a_frag(adjE, mt * 16, ks * 32, N_, lane);
                v16bf bb = b_fragT(s_ptmpT, ks * 32, nt * 16, N_, lane);
                acc2[i] = wmma_bf16(a, bb, acc2[i]);
            }
        }
        __syncthreads();
    }
#pragma unroll
    for (int i = 0; i < 2; ++i) {
        const int t = wave * 2 + i, mt = t >> 2, nt = t & 3;
        v16bf a  = a_frag(s_node, mt * 16, 0, F_, lane);
        v16bf bb = b_fragT(wp + WS_V2T, 0, nt * 16, F_, lane);
        acc2[i] = wmma_bf16(a, bb, acc2[i]);
        const int   mh8  = (lane >> 4) << 3;
        const float bias = c2[nt * 16 + n16];
        unsigned short* dst = s_ann2 + (mt * 16 + mh8) * AN2 + nt * 16 + n16;
#pragma unroll
        for (int v = 0; v < 8; ++v) dst[v * AN2] = f2bfu(fast_tanh(acc2[i][v] + bias));
    }
    __syncthreads();

    // ---------------- Phase 3: gates + node reduction ----------------
    // Gi/Gj: 64x128 from ann2(64x96); 32 tiles, 4 per wave; K = 96 (3 ksteps)
#pragma unroll
    for (int i = 0; i < 4; ++i) {
        const int t = wave * 4 + i, mt = t >> 3, nt = t & 7;
        v8f gi = {}, gj = {};
#pragma unroll
        for (int ks = 0; ks < 3; ++ks) {
            v16bf a   = a_frag(s_ann2, mt * 16, ks * 32, AN2, lane);
            v16bf bwi = b_fragT(wp + WS_WIT, ks * 32, nt * 16, AN2, lane);
            gi = wmma_bf16(a, bwi, gi);
            v16bf bwj = b_fragT(wp + WS_WJT, ks * 32, nt * 16, AN2, lane);
            gj = wmma_bf16(a, bwj, gj);
        }
        const int u = nt * 16 + n16;
        const float bbi = bi[u], bbj = bj[u];
        float partial = 0.0f;
#pragma unroll
        for (int v = 0; v < 8; ++v) {
            partial += fast_sigmoid(gi[v] + bbi) * fast_tanh(gj[v] + bbj);
        }
        atomicAdd(&s_red[u], partial);   // ds_add_f32
    }
    __syncthreads();

    if (tid < UA_) out[(size_t)b * UA_ + tid] = fast_tanh(s_red[tid]);
}

extern "C" void kernel_launch(void* const* d_in, const int* in_sizes, int n_in,
                              void* d_out, int out_size, void* d_ws, size_t ws_size,
                              hipStream_t stream) {
    (void)in_sizes; (void)n_in; (void)ws_size; (void)out_size;
    const float* adjacency = (const float*)d_in[0];
    /* d_in[1] = hidden (rank-2, zeros) -> unused by reference path */
    const float* node = (const float*)d_in[2];
    const float* W1 = (const float*)d_in[3];
    const float* b1 = (const float*)d_in[4];
    const float* V1 = (const float*)d_in[5];
    const float* c1 = (const float*)d_in[6];
    const float* W2 = (const float*)d_in[7];
    const float* b2 = (const float*)d_in[8];
    const float* V2 = (const float*)d_in[9];
    const float* c2 = (const float*)d_in[10];
    const float* Wi = (const float*)d_in[11];
    const float* bi = (const float*)d_in[12];
    const float* Wj = (const float*)d_in[13];
    const float* bj = (const float*)d_in[14];
    float* out = (float*)d_out;
    unsigned short* wp = (unsigned short*)d_ws;   // needs WS_END*2 = 176128 bytes

    // Pack weights to bf16 [N][K] layout (cheap; re-done every call for determinism)
    pack_weights_kernel<<<(WS_END + 255) / 256, 256, 0, stream>>>(W1, V1, W2, V2, Wi, Wj, wp);

    const size_t smem =
        (size_t)(NE * N_ * N_ + N_ * F_ + N_ * AN1 + N_ * AN2 + U1_ * N_) * sizeof(unsigned short)
        + UA_ * sizeof(float);   // 86528 bytes

    hipFuncSetAttribute((const void*)encoder_ggnn_kernel,
                        hipFuncAttributeMaxDynamicSharedMemorySize, (int)smem);

    encoder_ggnn_kernel<<<NB, 256, smem, stream>>>(adjacency, node, wp,
                                                   b1, c1, b2, c2, bi, bj, out);
}
